// Attention2d_75144747811148
// MI455X (gfx1250) — matrix operand order repaired and checked by reading, not compiled, not run on hardware
//
#include <hip/hip_runtime.h>

// ---------------- types / constants ----------------
typedef __attribute__((ext_vector_type(16))) __bf16 v16bf;
typedef __attribute__((ext_vector_type(8)))  float  v8f;
typedef unsigned short bf16_t;   // bf16 storage (bit pattern)

#define NB    8       // batch
#define NC    512     // channels
#define NN    1024    // tokens (H*W)
#define NHD   8       // heads
#define NCH   64      // channels per head
#define NG    32      // groupnorm groups
#define CPG   16      // channels per group
#define EPSV  1e-5f
#define QKS   0.35355339059327379f   // 64^-0.25, folded into q and k

__device__ __forceinline__ bf16_t f2bf(float f) {
  union { float f; unsigned u; } x; x.f = f;
  unsigned r = x.u + 0x7FFFu + ((x.u >> 16) & 1u);   // round-to-nearest-even
  return (bf16_t)(r >> 16);
}

// Async DMA: 16 bytes per lane, global -> LDS, tracked by ASYNCcnt (no VGPR data).
__device__ __forceinline__ void async_ld16(bf16_t* lds_dst, const bf16_t* g_src) {
  unsigned       loff  = (unsigned)(uintptr_t)lds_dst;           // LDS byte address (low 32 bits)
  unsigned long long ga = (unsigned long long)(uintptr_t)g_src;
  asm volatile("global_load_async_to_lds_b128 %0, %1, off"
               :: "v"(loff), "v"(ga) : "memory");
}

// Load a 16-element bf16 fragment as two 16B chunks (global or LDS source).
__device__ __forceinline__ v16bf ld_frag(const bf16_t* p0, const bf16_t* p1) {
  v16bf f;
  ((float4*)&f)[0] = *(const float4*)p0;
  ((float4*)&f)[1] = *(const float4*)p1;
  return f;
}
// A operand (16x32, bf16) from row-major [M][ld]; base pre-offset to (row0,k0).
__device__ __forceinline__ v16bf frag_a(const bf16_t* base, int ld, int lane) {
  const bf16_t* p = base + (lane & 15) * ld + (lane >> 4) * 8;
  return ld_frag(p, p + 16);
}
// B operand (32x16, bf16) from row-major [N][ld] (K contiguous); base pre-offset to (n0,k0).
__device__ __forceinline__ v16bf frag_b(const bf16_t* base, int ld, int lane) {
  const bf16_t* p = base + (lane & 15) * ld + (lane >> 4) * 8;
  return ld_frag(p, p + 16);
}
#define WMMA(a,b,c) __builtin_amdgcn_wmma_f32_16x16x32_bf16(false,(a),false,(b),(short)0,(c),false,false)

__device__ __forceinline__ float rmax16(float v) {
  #pragma unroll
  for (int o = 1; o < 16; o <<= 1) v = fmaxf(v, __shfl_xor(v, o));
  return v;
}
__device__ __forceinline__ float rsum16(float v) {
  #pragma unroll
  for (int o = 1; o < 16; o <<= 1) v += __shfl_xor(v, o);
  return v;
}

// ---------------- prep: weight conversion ----------------
__global__ void k_cvt_wqkv(const float* __restrict__ w, bf16_t* __restrict__ o, int n) {
  int i = blockIdx.x * 256 + threadIdx.x;
  if (i < n) {
    float v = w[i];
    if (i < 1024 * NC) v *= QKS;          // q and k rows: fold in ch^-0.25
    o[i] = f2bf(v);
  }
}
__global__ void k_cvt(const float* __restrict__ w, bf16_t* __restrict__ o, int n) {
  int i = blockIdx.x * 256 + threadIdx.x;
  if (i < n) o[i] = f2bf(w[i]);
}
__global__ void k_bias_qkv(const float* __restrict__ b, float* __restrict__ o) {
  int i = blockIdx.x * 256 + threadIdx.x;
  if (i < 3 * NC) o[i] = b[i] * (i < 1024 ? QKS : 1.0f);
}

// ---------------- GroupNorm1 -> hn bf16 [B][N][C] (token-major) ----------------
__global__ __launch_bounds__(256) void k_gn1(const float* __restrict__ x,
                                             const float* __restrict__ sc,
                                             const float* __restrict__ bi,
                                             bf16_t* __restrict__ hn) {
  int bg = blockIdx.x, b = bg / NG, g = bg % NG;
  const float* xp = x + (size_t)(b * NC + g * CPG) * NN;   // 16 ch x 1024 tok
  int tid = threadIdx.x;
  float s = 0.f, ss = 0.f;
  for (int i = tid; i < CPG * NN; i += 256) { float v = xp[i]; s += v; ss += v * v; }
  __shared__ float rs[256], rq[256];
  rs[tid] = s; rq[tid] = ss; __syncthreads();
  for (int o = 128; o; o >>= 1) {
    if (tid < o) { rs[tid] += rs[tid + o]; rq[tid] += rq[tid + o]; }
    __syncthreads();
  }
  float mu  = rs[0] * (1.0f / (CPG * NN));
  float var = rq[0] * (1.0f / (CPG * NN)) - mu * mu;
  float inv = rsqrtf(var + EPSV);
  for (int i = tid; i < CPG * NN; i += 256) {
    int cl = i >> 10, t = i & (NN - 1);
    int cc = g * CPG + cl;
    float v = (xp[i] - mu) * inv * sc[cc] + bi[cc];
    hn[((size_t)b * NN + t) * NC + cc] = f2bf(v);
  }
}

// ---------------- QKV GEMM with async-LDS staged B tiles ----------------
// grid (12, 16, B), block 256 (8 waves): wave -> 16(M) x 64(N) strip; B tile shared by block.
__global__ __launch_bounds__(256) void k_qkv(const bf16_t* __restrict__ hn,
                                             const bf16_t* __restrict__ wq,
                                             const float* __restrict__ bq,
                                             bf16_t* __restrict__ q,
                                             bf16_t* __restrict__ k,
                                             bf16_t* __restrict__ v) {
  __shared__ __align__(16) bf16_t bS[2][64 * 32];   // 64 tok x 32 k, double buffered (8 KB)
  int tid  = threadIdx.x;
  int lane = tid & 31, wv = tid >> 5;
  int half = lane >> 4, l16 = lane & 15;
  int b  = blockIdx.z;
  int m0 = blockIdx.x * 128 + wv * 16;       // output row (o) tile
  int t0 = blockIdx.y * 64;                  // token tile
  const bf16_t* hb = hn + (size_t)b * NN * NC;

  int srow = tid >> 2, sch = (tid & 3) * 8;  // staging map: 64 rows x 4 chunks of 8 elems
  // prologue: stage k0 = 0 into buffer 0
  async_ld16(&bS[0][srow * 32 + sch], hb + (size_t)(t0 + srow) * NC + 0 + sch);

  v8f acc[4] = {};
  for (int kt = 0; kt < 16; ++kt) {
    int k0 = kt * 32, cur = kt & 1;
    if (kt < 15) {
      async_ld16(&bS[cur ^ 1][srow * 32 + sch], hb + (size_t)(t0 + srow) * NC + (k0 + 32) + sch);
      asm volatile("s_wait_asynccnt 0x1" ::: "memory");   // previous stage done, prefetch in flight
    } else {
      asm volatile("s_wait_asynccnt 0x0" ::: "memory");
    }
    __syncthreads();
    v16bf af = frag_a(wq + (size_t)m0 * NC + k0, NC, lane);
    #pragma unroll
    for (int j = 0; j < 4; ++j) {
      v16bf bf_ = frag_b(&bS[cur][j * 16 * 32], 32, lane);
      acc[j] = WMMA(af, bf_, acc[j]);
    }
    __syncthreads();
  }

  int mbase = m0 + half * 8;
  #pragma unroll
  for (int j = 0; j < 4; ++j) {
    int t = t0 + j * 16 + l16;
    #pragma unroll
    for (int r = 0; r < 8; ++r) {
      int o = mbase + r;
      float val = acc[j][r] + bq[o];
      if (m0 < 512) {                        // q: [bh][t][c']
        int h = o >> 6, cc = o & 63;
        q[(((size_t)b * NHD + h) * NN + t) * NCH + cc] = f2bf(val);
      } else if (m0 < 1024) {                // k: [bh][s][c']
        int o2 = o - 512; int h = o2 >> 6, cc = o2 & 63;
        k[(((size_t)b * NHD + h) * NN + t) * NCH + cc] = f2bf(val);
      } else {                               // v: [bh][c'][s]
        int o2 = o - 1024; int h = o2 >> 6, cc = o2 & 63;
        v[(((size_t)b * NHD + h) * NCH + cc) * NN + t] = f2bf(val);
      }
    }
  }
}

// ---------------- Flash attention with async-LDS staged K/V tiles ----------------
// grid (8, 64), block 256 (8 waves). K/V s-block tiles shared by all 8 waves.
__global__ __launch_bounds__(256) void k_attn(const bf16_t* __restrict__ q,
                                              const bf16_t* __restrict__ k,
                                              const bf16_t* __restrict__ v,
                                              bf16_t* __restrict__ a) {
  __shared__ __align__(16) bf16_t kS[2][32 * 64];   // 32 s x 64 ch  (8 KB)
  __shared__ __align__(16) bf16_t vS[2][64 * 32];   // 64 ch x 32 s  (8 KB)
  __shared__ __align__(16) bf16_t Pb[8][16][32];    // per-wave P bounce (8 KB)
  int tid  = threadIdx.x;
  int lane = tid & 31, wv = tid >> 5;
  int half = lane >> 4, l16 = lane & 15;
  int bh = blockIdx.y, b = bh >> 3, h = bh & 7;
  const bf16_t* qb = q + (size_t)bh * NN * NCH;
  const bf16_t* kb = k + (size_t)bh * NN * NCH;
  const bf16_t* vb = v + (size_t)bh * NCH * NN;
  int t0 = blockIdx.x * 128 + wv * 16;

  v16bf qa0 = frag_a(qb + (size_t)t0 * NCH + 0,  NCH, lane);
  v16bf qa1 = frag_a(qb + (size_t)t0 * NCH + 32, NCH, lane);

  int vrow = tid >> 2, vch = (tid & 3) * 8;        // v staging: 64 rows x 4 chunks
  // prologue: stage s0 = 0 into buffer 0
  async_ld16(&kS[0][tid * 8], kb + 0 * NCH + tid * 8);                // contiguous 4 KB
  async_ld16(&vS[0][vrow * 32 + vch], vb + (size_t)vrow * NN + 0 + vch);

  float mrun[8], lrun[8];
  #pragma unroll
  for (int r = 0; r < 8; ++r) { mrun[r] = -1e30f; lrun[r] = 0.f; }
  v8f oacc[4] = {};

  for (int it = 0; it < NN / 32; ++it) {
    int s0 = it * 32, cur = it & 1;
    if (it < NN / 32 - 1) {
      int sn = s0 + 32;
      async_ld16(&kS[cur ^ 1][tid * 8], kb + (size_t)sn * NCH + tid * 8);
      async_ld16(&vS[cur ^ 1][vrow * 32 + vch], vb + (size_t)vrow * NN + sn + vch);
      asm volatile("s_wait_asynccnt 0x2" ::: "memory");  // current stage (2 instrs) complete
    } else {
      asm volatile("s_wait_asynccnt 0x0" ::: "memory");
    }
    __syncthreads();

    const bf16_t* kc = &kS[cur][0];
    const bf16_t* vc = &vS[cur][0];
    v16bf kb0 = frag_b(kc + 0,            NCH, lane);    // s rows 0..15,  K 0..31
    v16bf kb1 = frag_b(kc + 32,           NCH, lane);    // s rows 0..15,  K 32..63
    v16bf kb2 = frag_b(kc + 16 * NCH,     NCH, lane);    // s rows 16..31, K 0..31
    v16bf kb3 = frag_b(kc + 16 * NCH + 32, NCH, lane);
    v8f sA = {}; sA = WMMA(qa0, kb0, sA); sA = WMMA(qa1, kb1, sA);
    v8f sB = {}; sB = WMMA(qa0, kb2, sB); sB = WMMA(qa1, kb3, sB);

    float alpha[8];
    #pragma unroll
    for (int r = 0; r < 8; ++r) {
      float mt = rmax16(fmaxf(sA[r], sB[r]));
      float mn = fmaxf(mrun[r], mt);
      float pa = __expf(sA[r] - mn);
      float pb = __expf(sB[r] - mn);
      float rsum = rsum16(pa + pb);
      alpha[r] = __expf(mrun[r] - mn);
      lrun[r]  = lrun[r] * alpha[r] + rsum;
      mrun[r]  = mn;
      Pb[wv][r + 8 * half][l16]      = f2bf(pa);
      Pb[wv][r + 8 * half][16 + l16] = f2bf(pb);
    }
    #pragma unroll
    for (int jc = 0; jc < 4; ++jc)
      #pragma unroll
      for (int r = 0; r < 8; ++r) oacc[jc][r] *= alpha[r];

    asm volatile("s_wait_dscnt 0" ::: "memory");   // P tile visible for cross-lane LDS read
    v16bf pf = frag_a(&Pb[wv][0][0], 32, lane);    // A operand: 16 rows x K=32 (s-block)
    #pragma unroll
    for (int jc = 0; jc < 4; ++jc) {
      v16bf vf = frag_b(vc + (size_t)(jc * 16) * 32, 32, lane);
      oacc[jc] = WMMA(pf, vf, oacc[jc]);
    }
    __syncthreads();
  }

  #pragma unroll
  for (int jc = 0; jc < 4; ++jc) {
    int cc = h * NCH + jc * 16 + l16;
    #pragma unroll
    for (int r = 0; r < 8; ++r) {
      int t = t0 + r + 8 * half;
      float val = oacc[jc][r] / lrun[r];
      a[((size_t)b * NN + t) * NC + cc] = f2bf(val);
    }
  }
}

// ---------------- Proj GEMM with async-LDS staged B tiles ----------------
// grid (4, 16, B), block 256 (8 waves).
__global__ __launch_bounds__(256) void k_proj(const bf16_t* __restrict__ a,
                                              const bf16_t* __restrict__ wp,
                                              const float* __restrict__ bp,
                                              float* __restrict__ y) {
  __shared__ __align__(16) bf16_t bS[2][64 * 32];
  int tid  = threadIdx.x;
  int lane = tid & 31, wv = tid >> 5;
  int half = lane >> 4, l16 = lane & 15;
  int b  = blockIdx.z;
  int m0 = blockIdx.x * 128 + wv * 16;
  int t0 = blockIdx.y * 64;
  const bf16_t* ab = a + (size_t)b * NN * NC;

  int srow = tid >> 2, sch = (tid & 3) * 8;
  async_ld16(&bS[0][srow * 32 + sch], ab + (size_t)(t0 + srow) * NC + 0 + sch);

  v8f acc[4] = {};
  for (int kt = 0; kt < 16; ++kt) {
    int k0 = kt * 32, cur = kt & 1;
    if (kt < 15) {
      async_ld16(&bS[cur ^ 1][srow * 32 + sch], ab + (size_t)(t0 + srow) * NC + (k0 + 32) + sch);
      asm volatile("s_wait_asynccnt 0x1" ::: "memory");
    } else {
      asm volatile("s_wait_asynccnt 0x0" ::: "memory");
    }
    __syncthreads();
    v16bf af = frag_a(wp + (size_t)m0 * NC + k0, NC, lane);
    #pragma unroll
    for (int j = 0; j < 4; ++j) {
      v16bf bf_ = frag_b(&bS[cur][j * 16 * 32], 32, lane);
      acc[j] = WMMA(af, bf_, acc[j]);
    }
    __syncthreads();
  }

  int mbase = m0 + half * 8;
  #pragma unroll
  for (int j = 0; j < 4; ++j) {
    int t = t0 + j * 16 + l16;
    #pragma unroll
    for (int r = 0; r < 8; ++r) {
      int o = mbase + r;
      y[((size_t)b * NC + o) * NN + t] = acc[j][r] + bp[o];
    }
  }
}

// ---------------- GroupNorm2 + residual ----------------
__global__ __launch_bounds__(256) void k_gn2res(const float* __restrict__ y,
                                                const float* __restrict__ x,
                                                const float* __restrict__ sc,
                                                const float* __restrict__ bi,
                                                float* __restrict__ out) {
  int bg = blockIdx.x, b = bg / NG, g = bg % NG;
  size_t cb = (size_t)(b * NC + g * CPG) * NN;
  const float* yp = y + cb;
  int tid = threadIdx.x;
  float s = 0.f, ss = 0.f;
  for (int i = tid; i < CPG * NN; i += 256) { float v = yp[i]; s += v; ss += v * v; }
  __shared__ float rs[256], rq[256];
  rs[tid] = s; rq[tid] = ss; __syncthreads();
  for (int o = 128; o; o >>= 1) {
    if (tid < o) { rs[tid] += rs[tid + o]; rq[tid] += rq[tid + o]; }
    __syncthreads();
  }
  float mu  = rs[0] * (1.0f / (CPG * NN));
  float var = rq[0] * (1.0f / (CPG * NN)) - mu * mu;
  float inv = rsqrtf(var + EPSV);
  for (int i = tid; i < CPG * NN; i += 256) {
    int cc = g * CPG + (i >> 10);
    out[cb + i] = x[cb + i] + (yp[i] - mu) * inv * sc[cc] + bi[cc];
  }
}

// ---------------- host launcher ----------------
extern "C" void kernel_launch(void* const* d_in, const int* in_sizes, int n_in,
                              void* d_out, int out_size, void* d_ws, size_t ws_size,
                              hipStream_t stream) {
  const float* x    = (const float*)d_in[0];
  const float* g1s  = (const float*)d_in[1];
  const float* g1b  = (const float*)d_in[2];
  const float* wqkv = (const float*)d_in[3];
  const float* bqkv = (const float*)d_in[4];
  const float* wprj = (const float*)d_in[5];
  const float* bprj = (const float*)d_in[6];
  const float* g2s  = (const float*)d_in[7];
  const float* g2b  = (const float*)d_in[8];
  float* out = (float*)d_out;

  char* ws = (char*)d_ws;
  size_t off = 0;
  bf16_t* wq_b = (bf16_t*)(ws + off); off += (size_t)3 * NC * NC * 2;   // 1.5 MB
  bf16_t* wp_b = (bf16_t*)(ws + off); off += (size_t)NC * NC * 2;       // 0.5 MB
  float*  bq_s = (float*) (ws + off); off += 2048 * 4;                  // scaled qkv bias
  bf16_t* hn   = (bf16_t*)(ws + off); off += (size_t)NB * NN * NC * 2;  // 8 MB
  bf16_t* qb   = (bf16_t*)(ws + off); off += (size_t)NB * NN * NC * 2;  // 8 MB
  bf16_t* kb   = (bf16_t*)(ws + off); off += (size_t)NB * NN * NC * 2;  // 8 MB
  bf16_t* vb   = (bf16_t*)(ws + off); off += (size_t)NB * NN * NC * 2;  // 8 MB
  bf16_t* ab   = (bf16_t*)(ws + off); off += (size_t)NB * NN * NC * 2;  // 8 MB
  float*  yb   = (float*) (ws + off); off += (size_t)NB * NC * NN * 4;  // 16 MB

  // weight prep
  k_cvt_wqkv<<<(3 * NC * NC + 255) / 256, 256, 0, stream>>>(wqkv, wq_b, 3 * NC * NC);
  k_cvt     <<<(NC * NC + 255) / 256,     256, 0, stream>>>(wprj, wp_b, NC * NC);
  k_bias_qkv<<<(3 * NC + 255) / 256,      256, 0, stream>>>(bqkv, bq_s);

  // pipeline
  k_gn1 <<<NB * NG, 256, 0, stream>>>(x, g1s, g1b, hn);
  k_qkv <<<dim3(12, 16, NB), 256, 0, stream>>>(hn, wq_b, bq_s, qb, kb, vb);
  k_attn<<<dim3(NN / 128, NB * NHD), 256, 0, stream>>>(qb, kb, vb, ab);
  k_proj<<<dim3(4, 16, NB), 256, 0, stream>>>(ab, wp_b, bprj, yb);
  k_gn2res<<<NB * NG, 256, 0, stream>>>(yb, x, g2s, g2b, out);

  (void)in_sizes; (void)n_in; (void)out_size; (void)ws_size;
}
